// DotAttention_48859547959803
// MI455X (gfx1250) — matrix operand order repaired and checked by reading, not compiled, not run on hardware
//
#include <hip/hip_runtime.h>

// ---- problem constants (from reference) ----
#define Bn 16
#define Sn 8192
#define Qn 128
#define Dn 256

// ---- plain (trivially-copyable) vector types ----
typedef __attribute__((ext_vector_type(16))) __bf16       v16bf;
typedef __attribute__((ext_vector_type(8)))  __bf16       bf16x8;
typedef __attribute__((ext_vector_type(4)))  __bf16       bf16x4;
typedef __attribute__((ext_vector_type(8)))  float        v8f;
typedef __attribute__((ext_vector_type(8)))  float        f32x8;
typedef __attribute__((ext_vector_type(4)))  float        f32x4;
typedef __attribute__((ext_vector_type(4)))  unsigned int u32x4;
typedef __attribute__((ext_vector_type(2)))  unsigned int u32x2;

union UB {              // LDS bits -> WMMA B operand (no conversion needed)
    v16bf v;
    u32x4 q[2];
    u32x2 d[4];
};

// Build WMMA A operand from two contiguous 8-float chunks (packed cvt path).
static __device__ __forceinline__ v16bf mk_a(const float* pA, const float* pB) {
    f32x8 fa = *(const f32x8*)pA;      // 2x global_load_b128
    f32x8 fb = *(const f32x8*)pB;
    bf16x8 ha = __builtin_convertvector(fa, bf16x8);   // 4x v_cvt_pk_bf16_f32
    bf16x8 hb = __builtin_convertvector(fb, bf16x8);
    return __builtin_shufflevector(ha, hb,
        0, 1, 2, 3, 4, 5, 6, 7, 8, 9, 10, 11, 12, 13, 14, 15);
}

// 4 floats -> 4 packed bf16 (two dwords), for row-major LDS staging.
static __device__ __forceinline__ u32x2 pk4(f32x4 f) {
    bf16x4 h = __builtin_convertvector(f, bf16x4);     // 2x v_cvt_pk_bf16_f32
    return __builtin_bit_cast(u32x2, h);
}

// Interleave two row-fragments (same d-range, s and s+1) into 4 bf16-pair
// dwords {f0.x|f1.x, f0.y|f1.y, ...} for transposed LDS staging.
static __device__ __forceinline__ u32x4 pk_pairs(f32x4 f0, f32x4 f1) {
    f32x8 z = __builtin_shufflevector(f0, f1, 0, 4, 1, 5, 2, 6, 3, 7);
    bf16x8 h = __builtin_convertvector(z, bf16x8);     // 4x v_cvt_pk_bf16_f32
    return __builtin_bit_cast(u32x4, h);
}

// ===================================================================
// K1: scores[q,b,s] = sum_d query[q,b,d]*context[b,s,d], masked -> attn (raw)
//   grid = (S/64, B), 256 threads (8 waves = 8 q-tiles of 16)
//   context chunk [64s x 256d] staged once in LDS as bf16 (row-major)
// ===================================================================
#define K1_ROWS 64
#define K1_RB   264   // LDS row stride in shorts (256 + 8 pad -> bank spread)

__global__ __launch_bounds__(256) void k1_scores(
    const float* __restrict__ ctx, const float* __restrict__ qry,
    const unsigned char* __restrict__ mask, float* __restrict__ attn)
{
    __shared__ __align__(16) unsigned short lds[K1_ROWS * K1_RB];
    const int b    = blockIdx.y;
    const int s0g  = blockIdx.x * K1_ROWS;
    const int tid  = threadIdx.x;
    const int w    = tid >> 5;
    const int lane = tid & 31;
    const int m    = lane & 15;
    const int g    = lane >> 4;
    const int q0   = w * 16;

    // ---- preload A: query tile (16q x 256d) in WMMA-A bf16 layout ----
    v16bf a[8];
    {
        const float* qrow = qry + ((size_t)(q0 + m) * Bn + b) * Dn;
#pragma unroll
        for (int kb = 0; kb < 8; ++kb)
            a[kb] = mk_a(qrow + kb * 32 + g * 16, qrow + kb * 32 + g * 16 + 8);
    }

    // ---- stage context chunk -> LDS bf16, natural [s][d] layout ----
#pragma unroll 4
    for (int it = 0; it < 16; ++it) {
        int id = tid + 256 * it;
        int s  = id >> 6;
        int d4 = (id & 63) << 2;
        f32x4 f = *(const f32x4*)(ctx + ((size_t)b * Sn + s0g + s) * Dn + d4);
        *(u32x2*)&lds[s * K1_RB + d4] = pk4(f);   // 8B store, conflict-free
    }
    __syncthreads();

    const int kbase = g * 16;  // B layout: lanes 0-15 -> K 0..15, 16-31 -> 16..31
#pragma unroll
    for (int st = 0; st < 4; ++st) {
        const int srow = st * 16 + m;     // lane's B column = context row
        v8f c = {};
#pragma unroll
        for (int kb = 0; kb < 8; ++kb) {
            const u32x4* p = (const u32x4*)&lds[srow * K1_RB + kb * 32 + kbase];
            UB ub; ub.q[0] = p[0]; ub.q[1] = p[1];
            c = __builtin_amdgcn_wmma_f32_16x16x32_bf16(
                    false, a[kb], false, ub.v, (short)0, c, false, false);
        }
        int sg = s0g + st * 16 + m;
        bool msk = mask[(size_t)b * Sn + sg] != 0;
#pragma unroll
        for (int r = 0; r < 8; ++r) {
            int qrow = q0 + g * 8 + r;    // D layout: M = 8*(lane/16) + vgpr
            attn[((size_t)qrow * Bn + b) * Sn + sg] = msk ? -INFINITY : c[r];
        }
    }
}

// ===================================================================
// K2: in-place row softmax over attn rows, + copy query into output[:,:,256:512]
//   grid = Q*B blocks (one row each), 256 threads
// ===================================================================
__global__ __launch_bounds__(256) void k2_softmax(
    const float* __restrict__ qry, float* __restrict__ out,
    float* __restrict__ attn)
{
    const int rid = blockIdx.x;           // rid = q*B + b
    const int tid = threadIdx.x;
    float* row = attn + (size_t)rid * Sn;

    __shared__ float red[8];

    float mx = -INFINITY;
    for (int i = tid; i < Sn; i += 256) mx = fmaxf(mx, row[i]);
#pragma unroll
    for (int off = 16; off; off >>= 1) mx = fmaxf(mx, __shfl_xor(mx, off));
    if ((tid & 31) == 0) red[tid >> 5] = mx;
    __syncthreads();
    mx = red[0];
#pragma unroll
    for (int i = 1; i < 8; ++i) mx = fmaxf(mx, red[i]);
    __syncthreads();

    float sm = 0.f;
    for (int i = tid; i < Sn; i += 256) {
        float e = __expf(row[i] - mx);    // masked (-inf) -> 0
        row[i] = e;
        sm += e;
    }
#pragma unroll
    for (int off = 16; off; off >>= 1) sm += __shfl_xor(sm, off);
    if ((tid & 31) == 0) red[tid >> 5] = sm;
    __syncthreads();
    sm = 0.f;
#pragma unroll
    for (int i = 0; i < 8; ++i) sm += red[i];
    const float inv = 1.0f / sm;
    for (int i = tid; i < Sn; i += 256) row[i] *= inv;   // own elements only

    // concat second half: output[q,b,256+d] = query[q,b,d]
    out[(size_t)rid * (2 * Dn) + Dn + tid] = qry[(size_t)rid * Dn + tid];
}

// ===================================================================
// K3: read[q,b,d] = sum_s attn[q,b,s]*context[b,s,d] -> output[:,:,0:256]
//   grid = (Q/16, B), 256 threads; wave w owns d-tiles {2w, 2w+1}
//   context [32s x 256d] staged TRANSPOSED in LDS ([d][s], stride 36 shorts)
// ===================================================================
#define K3_RT 36   // transposed LDS row stride in shorts (32 + 4 pad)

__global__ __launch_bounds__(256) void k3_av(
    const float* __restrict__ ctx, const float* __restrict__ attn,
    float* __restrict__ out)
{
    __shared__ __align__(16) unsigned short ldsT[Dn * K3_RT];
    const int b    = blockIdx.y;
    const int q0   = blockIdx.x * 16;
    const int tid  = threadIdx.x;
    const int w    = tid >> 5;
    const int lane = tid & 31;
    const int m    = lane & 15;
    const int g    = lane >> 4;
    const int kbase = g * 16;
    const int d0    = w * 32;

    v8f c0 = {}, c1 = {};
    const float* prow = attn + ((size_t)(q0 + m) * Bn + b) * Sn;

    for (int sb = 0; sb < Sn; sb += 32) {
        // ---- stage context[b, sb..sb+31, :] transposed to LDS [d][s] ----
#pragma unroll
        for (int it = 0; it < 4; ++it) {
            int id = tid + 256 * it;
            int sp = id >> 6;                 // s-pair 0..15
            int d4 = (id & 63) << 2;
            const float* base = ctx + ((size_t)b * Sn + sb + sp * 2) * Dn + d4;
            f32x4 f0 = *(const f32x4*)(base);        // s = sb+2sp
            f32x4 f1 = *(const f32x4*)(base + Dn);   // s = sb+2sp+1
            u32x4 pz = pk_pairs(f0, f1);
            *(unsigned int*)&ldsT[(d4 + 0) * K3_RT + sp * 2] = pz.x;
            *(unsigned int*)&ldsT[(d4 + 1) * K3_RT + sp * 2] = pz.y;
            *(unsigned int*)&ldsT[(d4 + 2) * K3_RT + sp * 2] = pz.z;
            *(unsigned int*)&ldsT[(d4 + 3) * K3_RT + sp * 2] = pz.w;
        }
        __syncthreads();

        // ---- A = probability tile 16q x 32s (bf16) ----
        v16bf ua = mk_a(prow + sb + g * 16, prow + sb + g * 16 + 8);

        // ---- B tiles from transposed LDS: lane column = d0(+16)+m ----
        const u32x2* p0 = (const u32x2*)&ldsT[(d0 + m) * K3_RT + kbase];
        const u32x2* p1 = (const u32x2*)&ldsT[(d0 + 16 + m) * K3_RT + kbase];
        UB ub0, ub1;
#pragma unroll
        for (int i = 0; i < 4; ++i) { ub0.d[i] = p0[i]; ub1.d[i] = p1[i]; }

        c0 = __builtin_amdgcn_wmma_f32_16x16x32_bf16(
                 false, ua, false, ub0.v, (short)0, c0, false, false);
        c1 = __builtin_amdgcn_wmma_f32_16x16x32_bf16(
                 false, ua, false, ub1.v, (short)0, c1, false, false);
        __syncthreads();
    }

#pragma unroll
    for (int r = 0; r < 8; ++r) {
        int qrow = q0 + g * 8 + r;
        size_t o = ((size_t)qrow * Bn + b) * (size_t)(2 * Dn);
        out[o + d0 + m]      = c0[r];
        out[o + d0 + 16 + m] = c1[r];
    }
}

// ===================================================================
extern "C" void kernel_launch(void* const* d_in, const int* in_sizes, int n_in,
                              void* d_out, int out_size, void* d_ws, size_t ws_size,
                              hipStream_t stream) {
    (void)in_sizes; (void)n_in; (void)out_size; (void)d_ws; (void)ws_size;
    const float*         ctx  = (const float*)d_in[0];          // [B,S,D] f32
    const float*         qry  = (const float*)d_in[1];          // [Q,B,D] f32
    const unsigned char* mask = (const unsigned char*)d_in[2];  // [B,S] bool (1B)

    float* out  = (float*)d_out;                         // [Q,B,2D]
    float* attn = out + (size_t)Qn * Bn * 2 * Dn;        // [Q,B,S]

    k1_scores <<<dim3(Sn / K1_ROWS, Bn), 256, 0, stream>>>(ctx, qry, mask, attn);
    k2_softmax<<<dim3(Qn * Bn),          256, 0, stream>>>(qry, out, attn);
    k3_av     <<<dim3(Qn / 16, Bn),      256, 0, stream>>>(ctx, attn, out);
}
